// GraphAttentionLayer_37349035606779
// MI455X (gfx1250) — matrix operand order repaired and checked by reading, not compiled, not run on hardware
//
#include <hip/hip_runtime.h>
#include <hip/hip_bf16.h>
#include <cstdint>

#define N_NODES 8192
#define IN_FEAT 512
#define OUT_FEAT 64
#define ALPHA 0.2f
#define NEG_BIG -1.0e30f
#define TILE_COLS 128
#define NTILES (N_NODES / TILE_COLS)   // 64

typedef __attribute__((ext_vector_type(16))) __bf16 v16bf;
typedef __attribute__((ext_vector_type(8)))  float  v8f;
typedef __attribute__((ext_vector_type(4)))  unsigned int v4u;
typedef __attribute__((ext_vector_type(8)))  int v8i_;
typedef __attribute__((ext_vector_type(4)))  int v4i_;

// ---------------------------------------------------------------------------
// TDM: issue a 2D tile load (tile_d1 rows x tile_d0 elems) from a row-major
// tensor with row stride `stride_elems` into LDS at byte offset lds_off.
// D# packing per CDNA5 ISA (groups 0 and 1; groups 2/3
// zero for a 2D tensor). data_size_code: 0=1B,1=2B,2=4B,3=8B.
// This toolchain exposes the 6-arg builtin:
//   (uint32x4 g0, int32x8 g1, int32x4 g2, int32x4 g3, int32x8 extra, i32 cpol)
// ---------------------------------------------------------------------------
__device__ __forceinline__ void tdm_load_2d(unsigned lds_off, const void* gptr,
                                            unsigned tile_d0, unsigned tile_d1,
                                            unsigned long long stride_elems,
                                            unsigned data_size_code,
                                            unsigned tensor_d0, unsigned tensor_d1) {
  const unsigned long long ga = (unsigned long long)gptr;
  v4u g0;
  g0[0] = 1u;                                               // count=1, user mode
  g0[1] = lds_off;                                          // lds_addr
  g0[2] = (unsigned)(ga & 0xFFFFFFFFu);                     // global_addr[31:0]
  g0[3] = (unsigned)((ga >> 32) & 0x1FFFFFFu) | 0x80000000u; // [56:32] | type=2
  v8i_ g1;
  g1[0] = (int)(data_size_code << 16);                      // wg_mask=0, data_size
  g1[1] = (int)((tensor_d0 & 0xFFFFu) << 16);               // tensor_dim0[15:0]
  g1[2] = (int)((tensor_d0 >> 16) | ((tensor_d1 & 0xFFFFu) << 16));
  g1[3] = (int)((tensor_d1 >> 16) | ((tile_d0 & 0xFFFFu) << 16)); // tile_dim0
  g1[4] = (int)(tile_d1 & 0xFFFFu);                         // tile_dim1; tile_dim2=0
  g1[5] = (int)(stride_elems & 0xFFFFFFFFull);              // dim0_stride[31:0]
  g1[6] = (int)((stride_elems >> 32) & 0xFFFFull);          // dim0_stride[47:32]
  g1[7] = 0;
  v4i_ g2 = {0, 0, 0, 0};
  v4i_ g3 = {0, 0, 0, 0};
  v8i_ gx = {0, 0, 0, 0, 0, 0, 0, 0};
  __builtin_amdgcn_tensor_load_to_lds(g0, g1, g2, g3, gx, 0);
}

// ---------------------------------------------------------------------------
// Kernel 1: h = X @ W   (bf16 WMMA, f32 accumulate). One wave per 16x16 tile.
// ---------------------------------------------------------------------------
__global__ void __launch_bounds__(32)
gat_gemm_h(const float* __restrict__ X, const float* __restrict__ W,
           float* __restrict__ h, __bf16* __restrict__ hb) {
  const int tile = blockIdx.x;
  const int m0 = (tile >> 2) * 16;
  const int n0 = (tile & 3) * 16;
  const int lane = threadIdx.x;
  const int r = lane & 15;
  const int grp = lane >> 4;
  const int row = m0 + r;
  const int col = n0 + r;

  v8f acc = {};
  for (int k0 = 0; k0 < IN_FEAT; k0 += 32) {
    v16bf a, b;
#pragma unroll
    for (int i = 0; i < 16; ++i) {
      const int ka = k0 + (i & 7) + ((i >> 3) << 4) + (grp << 3);
      a[i] = (__bf16)X[(size_t)row * IN_FEAT + ka];
      const int kb = k0 + (i & 7) + ((i >> 3) << 4) + (grp << 3);
      b[i] = (__bf16)W[(size_t)kb * OUT_FEAT + col];
    }
    acc = __builtin_amdgcn_wmma_f32_16x16x32_bf16(false, a, false, b,
                                                  (short)0, acc, false, false);
  }
#pragma unroll
  for (int v = 0; v < 8; ++v) {
    const int orow = m0 + (grp << 3) + v;
    const float val = acc[v];
    h[(size_t)orow * OUT_FEAT + col] = val;
    hb[(size_t)orow * OUT_FEAT + col] = (__bf16)val;
  }
}

// ---------------------------------------------------------------------------
// Kernel 2: f_src = h @ a[:64], f_dst = h @ a[64:]. One thread per row.
// ---------------------------------------------------------------------------
__global__ void __launch_bounds__(256)
gat_fsd(const float* __restrict__ h, const float* __restrict__ a,
        float* __restrict__ fs, float* __restrict__ fd) {
  const int row = blockIdx.x * blockDim.x + threadIdx.x;
  if (row >= N_NODES) return;
  const float* hr = h + (size_t)row * OUT_FEAT;
  float s = 0.f, d = 0.f;
#pragma unroll 8
  for (int f = 0; f < OUT_FEAT; ++f) {
    const float hv = hr[f];
    s += hv * a[f];
    d += hv * a[OUT_FEAT + f];
  }
  fs[row] = s;
  fd[row] = d;
}

// ---------------------------------------------------------------------------
// Kernel 3: fused masked-score -> online-softmax -> att@h -> ELU with a
// depth-2 TDM double-buffer pipeline streaming adj (16x128 i32) and hb
// (128x64 bf16) tiles into LDS. One block per 16-row tile, 4 waves; wave w
// owns the 32-column sub-chunk w of each 128-column tile.
// ---------------------------------------------------------------------------
__global__ void __launch_bounds__(128)
gat_attn(const int* __restrict__ adj, const __bf16* __restrict__ hb,
         const float* __restrict__ fs, const float* __restrict__ fd,
         float* __restrict__ out) {
  __shared__ int    s_adj[2][16][TILE_COLS];         // 2 x 8 KB
  __shared__ __bf16 s_hb[2][TILE_COLS][OUT_FEAT];    // 2 x 16 KB
  __shared__ float  s_m[4][16];
  __shared__ float  s_l[4][16];
  __shared__ float  s_acc[4][16][OUT_FEAT];          // 16 KB

  const int m0 = blockIdx.x * 16;
  const int tid = threadIdx.x;
  const int w = tid >> 5;
  const int lane = tid & 31;
  const int r = lane & 15;
  const int grp = lane >> 4;
  const int row = m0 + r;

  const float fsr = fs[row];

  float m = NEG_BIG;
  float l = 0.f;
  v8f acc0 = {}, acc1 = {}, acc2 = {}, acc3 = {};

  // Prime the pipeline: tiles 0 and 1 (adj + hb each).
  if (w == 0) {
#pragma unroll
    for (int t = 0; t < 2; ++t) {
      tdm_load_2d((unsigned)(unsigned long long)(uintptr_t)&s_adj[t][0][0],
                  adj + (size_t)m0 * N_NODES + t * TILE_COLS,
                  TILE_COLS, 16, (unsigned long long)N_NODES,
                  2u, N_NODES, N_NODES);
      tdm_load_2d((unsigned)(unsigned long long)(uintptr_t)&s_hb[t][0][0],
                  hb + (size_t)(t * TILE_COLS) * OUT_FEAT,
                  OUT_FEAT, TILE_COLS, (unsigned long long)OUT_FEAT,
                  1u, OUT_FEAT, N_NODES);
    }
  }

  for (int t = 0; t < NTILES; ++t) {
    const int buf = t & 1;
    const int jc = t * TILE_COLS;

    // Tile t's two TDM loads are the oldest outstanding (in-order per wave).
    if (w == 0) __builtin_amdgcn_s_wait_tensorcnt(2);
    __syncthreads();

    // ---- scores for this wave's 32-column sub-chunk ----
    float sc[16];
    float lm = NEG_BIG;
#pragma unroll
    for (int i = 0; i < 16; ++i) {
      const int kk = (i & 7) + ((i >> 3) << 4) + (grp << 3);
      const int jl = (w << 5) + kk;             // column within tile
      const int av = s_adj[buf][r][jl];
      float e = fsr + fd[jc + jl];
      e = (e > 0.f) ? e : ALPHA * e;            // LeakyReLU
      const float s = (av > 0) ? e : NEG_BIG;
      sc[i] = s;
      lm = fmaxf(lm, s);
    }
    lm = fmaxf(lm, __shfl_xor(lm, 16, 32));     // lanes r / r+16 share a row
    const float mnew = fmaxf(m, lm);
    const float scale = __expf(m - mnew);
    l *= scale;
#pragma unroll
    for (int v = 0; v < 8; ++v) {
      acc0[v] *= scale; acc1[v] *= scale; acc2[v] *= scale; acc3[v] *= scale;
    }

    v16bf p;
    float lsum = 0.f;
#pragma unroll
    for (int i = 0; i < 16; ++i) {
      const float pv = (sc[i] > NEG_BIG) ? __expf(sc[i] - mnew) : 0.f;
      lsum += pv;
      p[i] = (__bf16)pv;
    }
    lsum += __shfl_xor(lsum, 16, 32);
    l += lsum;

    // ---- B operands from the LDS hb tile ----
    const __bf16* hbase = &s_hb[buf][w << 5][0] + r;
    v16bf b0, b1, b2, b3;
#pragma unroll
    for (int i = 0; i < 16; ++i) {
      const __bf16* hp = hbase + ((i & 7) + ((i >> 3) << 4) + (grp << 3)) * OUT_FEAT;
      b0[i] = hp[0];  b1[i] = hp[16];  b2[i] = hp[32];  b3[i] = hp[48];
    }
    acc0 = __builtin_amdgcn_wmma_f32_16x16x32_bf16(false, p, false, b0, (short)0, acc0, false, false);
    acc1 = __builtin_amdgcn_wmma_f32_16x16x32_bf16(false, p, false, b1, (short)0, acc1, false, false);
    acc2 = __builtin_amdgcn_wmma_f32_16x16x32_bf16(false, p, false, b2, (short)0, acc2, false, false);
    acc3 = __builtin_amdgcn_wmma_f32_16x16x32_bf16(false, p, false, b3, (short)0, acc3, false, false);
    m = mnew;

    __syncthreads();   // everyone done reading buf before it is overwritten

    if (w == 0 && t + 2 < NTILES) {
      const int tn = t + 2;
      tdm_load_2d((unsigned)(unsigned long long)(uintptr_t)&s_adj[buf][0][0],
                  adj + (size_t)m0 * N_NODES + (size_t)tn * TILE_COLS,
                  TILE_COLS, 16, (unsigned long long)N_NODES,
                  2u, N_NODES, N_NODES);
      tdm_load_2d((unsigned)(unsigned long long)(uintptr_t)&s_hb[buf][0][0],
                  hb + (size_t)(tn * TILE_COLS) * OUT_FEAT,
                  OUT_FEAT, TILE_COLS, (unsigned long long)OUT_FEAT,
                  1u, OUT_FEAT, N_NODES);
    }
  }

  // ---- dump per-wave partials & merge ----
  if (grp == 0) { s_m[w][r] = m; s_l[w][r] = l; }
#pragma unroll
  for (int v = 0; v < 8; ++v) {
    const int orow = (grp << 3) + v;
    s_acc[w][orow][ 0 + r] = acc0[v];
    s_acc[w][orow][16 + r] = acc1[v];
    s_acc[w][orow][32 + r] = acc2[v];
    s_acc[w][orow][48 + r] = acc3[v];
  }
  __syncthreads();

  for (int idx = tid; idx < 16 * OUT_FEAT; idx += 128) {
    const int rr = idx >> 6;
    const int f = idx & 63;
    const float M = fmaxf(fmaxf(s_m[0][rr], s_m[1][rr]),
                          fmaxf(s_m[2][rr], s_m[3][rr]));
    float L = 0.f, V = 0.f;
#pragma unroll
    for (int ww = 0; ww < 4; ++ww) {
      const float swt = __expf(s_m[ww][rr] - M);
      L += swt * s_l[ww][rr];
      V += swt * s_acc[ww][rr][f];
    }
    float o = V / fmaxf(L, 1e-30f);
    o = (o > 0.f) ? o : (__expf(o) - 1.f);      // ELU
    out[(size_t)(m0 + rr) * OUT_FEAT + f] = o;
  }
}

// ---------------------------------------------------------------------------
extern "C" void kernel_launch(void* const* d_in, const int* in_sizes, int n_in,
                              void* d_out, int out_size, void* d_ws, size_t ws_size,
                              hipStream_t stream) {
  (void)in_sizes; (void)n_in; (void)out_size; (void)ws_size;
  const float* X   = (const float*)d_in[0];  // [N, 512]
  const int*   adj = (const int*)d_in[1];    // [N, N]
  const float* W   = (const float*)d_in[2];  // [512, 64]
  const float* a   = (const float*)d_in[3];  // [128, 1]
  float* out = (float*)d_out;                // [N, 64]

  char* ws = (char*)d_ws;
  float*  h  = (float*)ws;                                          // 2 MB
  __bf16* hb = (__bf16*)(ws + (size_t)N_NODES * OUT_FEAT * 4);      // 1 MB
  float*  fs = (float*)(ws + (size_t)N_NODES * OUT_FEAT * 6);       // 32 KB
  float*  fd = fs + N_NODES;                                        // 32 KB

  gat_gemm_h<<<(N_NODES / 16) * (OUT_FEAT / 16), 32, 0, stream>>>(X, W, h, hb);
  gat_fsd<<<N_NODES / 256, 256, 0, stream>>>(h, a, fs, fd);
  gat_attn<<<N_NODES / 16, 128, 0, stream>>>(adj, hb, fs, fd, out);
}
